// ResidualGATBlock_77120432767649
// MI455X (gfx1250) — matrix operand order repaired and checked by reading, not compiled, not run on hardware
//
#include <hip/hip_runtime.h>
#include <math.h>

#define NN 20000
#define NE 320000
#define IND 256
#define HEADS 4
#define HDIM 64
#define OUTD 256
#define NEG_SLOPE 0.2f
#define LN_EPS 1e-5f
#define SM_EPS 1e-16f

typedef __attribute__((ext_vector_type(16))) __bf16 v16bf;
typedef __attribute__((ext_vector_type(8)))  float v8f;

__device__ __forceinline__ unsigned short f2bf(float f) {
    unsigned int u = __float_as_uint(f);
    u += 0x7FFFu + ((u >> 16) & 1u);   // round to nearest even
    return (unsigned short)(u >> 16);
}

__device__ __forceinline__ float atomicMaxF(float* addr, float val) {
    if (val >= 0.0f)
        return __int_as_float(atomicMax((int*)addr, __float_as_int(val)));
    else
        return __uint_as_float(atomicMin((unsigned int*)addr, __float_as_uint(val)));
}

// ---------------- kernel 1: x (f32) -> bf16 ----------------
__global__ void k_cvt_x(const float* __restrict__ x, unsigned short* __restrict__ xb) {
    long i = (long)blockIdx.x * blockDim.x + threadIdx.x;
    long n = (long)NN * IND;
    for (; i < n; i += (long)gridDim.x * blockDim.x) xb[i] = f2bf(x[i]);
}

// ---------------- kernel 2: W (f32, [k][m]) -> bf16 transposed [m][k] ----------------
__global__ void k_cvt_w(const float* __restrict__ Wl, const float* __restrict__ Wr,
                        unsigned short* __restrict__ Wlb, unsigned short* __restrict__ Wrb) {
    int i = blockIdx.x * blockDim.x + threadIdx.x;       // i over 256*256
    if (i >= IND * OUTD) return;
    int m = i / IND, k = i % IND;
    Wlb[m * IND + k] = f2bf(Wl[k * OUTD + m]);
    Wrb[m * IND + k] = f2bf(Wr[k * OUTD + m]);
}

// ---------------- kernel 3: WMMA GEMM: xl = x@Wl, xr = x@Wr (bf16 in, f32 out) ----------------
// one wave per 16x16 output tile; jobs = rowTiles(1250) * 32  (16 col tiles * 2 matrices)
__global__ void k_gemm(const unsigned short* __restrict__ xb,
                       const unsigned short* __restrict__ Wlb,
                       const unsigned short* __restrict__ Wrb,
                       float* __restrict__ xl, float* __restrict__ xr) {
    int wave = threadIdx.x >> 5;
    int lane = threadIdx.x & 31;
    int job  = blockIdx.x * (blockDim.x >> 5) + wave;
    if (job >= 1250 * 32) return;

    int rt    = job >> 5;          // row tile 0..1249
    int rem   = job & 31;
    int which = rem >> 4;          // 0 = W_l, 1 = W_r
    int ct    = rem & 15;          // col tile 0..15
    int r0 = rt * 16, c0 = ct * 16;

    int m16    = lane & 15;
    int kbaseA = (lane < 16) ? 0 : 8;    // A: 16x32 bf16 layout (ISA 7.12.2)
    int kbaseB = (lane < 16) ? 0 : 16;   // unused; B is loaded with kbaseA (same k order as A)

    const unsigned short* Arow = xb + (long)(r0 + m16) * IND;
    const unsigned short* Wt   = which ? Wrb : Wlb;
    const unsigned short* Brow = Wt + (long)(c0 + m16) * IND;   // row m of W^T = column m of W

    v8f acc = {};
    #pragma unroll
    for (int k = 0; k < IND; k += 32) {
        union { unsigned short u[16]; v16bf v; } a, b;
        #pragma unroll
        for (int e = 0; e < 8; ++e) {
            a.u[e]     = Arow[k + kbaseA + e];          // K = k+kbase   .. +7   (VGPR0-3)
            a.u[e + 8] = Arow[k + 16 + kbaseA + e];     // K = k+16+kbase.. +7   (VGPR4-7)
        }
        #pragma unroll
        for (int e = 0; e < 8; ++e) {
            b.u[e]     = Brow[k + kbaseA + e];          // K = k+kbase   .. +7 (same as A)
            b.u[e + 8] = Brow[k + 16 + kbaseA + e];     // K = k+16+kbase.. +7
        }
        acc = __builtin_amdgcn_wmma_f32_16x16x32_bf16(false, a.v, false, b.v,
                                                      (short)0, acc, false, false);
    }

    float* Out = which ? xr : xl;
    int mbase = (lane < 16) ? 0 : 8;
    #pragma unroll
    for (int r = 0; r < 8; ++r)
        Out[(long)(r0 + mbase + r) * OUTD + c0 + m16] = acc[r];
}

// ---------------- kernel 4: init accumulators (every launch) ----------------
__global__ void k_init(float* __restrict__ agg, float* __restrict__ node_max,
                       float* __restrict__ node_sum) {
    long i = (long)blockIdx.x * blockDim.x + threadIdx.x;
    long n = (long)NN * OUTD;
    for (long j = i; j < n; j += (long)gridDim.x * blockDim.x) agg[j] = 0.0f;
    long nh = (long)NN * HEADS;
    for (long j = i; j < nh; j += (long)gridDim.x * blockDim.x) {
        node_max[j] = -INFINITY;
        node_sum[j] = 0.0f;
    }
}

// ---------------- kernel 5: edge scores + segment max (one wave per edge) ----------------
__global__ void k_score(const int* __restrict__ ei, const float* __restrict__ xl,
                        const float* __restrict__ xr, const float* __restrict__ att,
                        float* __restrict__ score, float* __restrict__ node_max) {
    int wave = threadIdx.x >> 5;
    int lane = threadIdx.x & 31;
    int e = blockIdx.x * (blockDim.x >> 5) + wave;
    if (e >= NE) return;

    int head = lane >> 3;                 // 8 lanes per head
    // per-lane fixed att slice: att[head][ (lane&7)*8 + j ]
    float areg[8];
    #pragma unroll
    for (int j = 0; j < 8; ++j) areg[j] = att[head * HDIM + (lane & 7) * 8 + j];

    int src = ei[e];
    int dst = ei[NE + e];
    const float* pl = xl + (long)src * OUTD + lane * 8;
    const float* pr = xr + (long)dst * OUTD + lane * 8;

    float partial = 0.0f;
    #pragma unroll
    for (int j = 0; j < 8; ++j) {
        float v = pl[j] + pr[j];
        v = (v >= 0.0f) ? v : NEG_SLOPE * v;
        partial += v * areg[j];
    }
    // reduce within groups of 8 lanes
    partial += __shfl_xor(partial, 1, 32);
    partial += __shfl_xor(partial, 2, 32);
    partial += __shfl_xor(partial, 4, 32);

    if ((lane & 7) == 0) {
        score[(long)e * HEADS + head] = partial;
        atomicMaxF(&node_max[(long)dst * HEADS + head], partial);
    }
}

// ---------------- kernel 6: exp + segment sum (one thread per edge-head, in place) ------
__global__ void k_expsum(const int* __restrict__ ei, float* __restrict__ score,
                         const float* __restrict__ node_max, float* __restrict__ node_sum) {
    long i = (long)blockIdx.x * blockDim.x + threadIdx.x;
    if (i >= (long)NE * HEADS) return;
    int e = (int)(i >> 2);
    int h = (int)(i & 3);
    int dst = ei[NE + e];
    float m = node_max[(long)dst * HEADS + h];
    float ex = __expf(score[i] - m);
    score[i] = ex;                                  // reuse buffer for ex
    atomicAdd(&node_sum[(long)dst * HEADS + h], ex);
}

// ---------------- kernel 7: weighted scatter-add aggregation (one wave per edge) --------
__global__ void k_agg(const int* __restrict__ ei, const float* __restrict__ xl,
                      const float* __restrict__ exbuf, const float* __restrict__ node_sum,
                      float* __restrict__ agg) {
    int wave = threadIdx.x >> 5;
    int lane = threadIdx.x & 31;
    int e = blockIdx.x * (blockDim.x >> 5) + wave;
    if (e >= NE) return;

    int src = ei[e];
    int dst = ei[NE + e];
    int head = lane >> 3;
    float ex    = exbuf[(long)e * HEADS + head];
    float denom = node_sum[(long)dst * HEADS + head] + SM_EPS;
    float alpha = ex / denom;

    const float* pl = xl + (long)src * OUTD + lane * 8;
    float* pd = agg + (long)dst * OUTD + lane * 8;
    #pragma unroll
    for (int j = 0; j < 8; ++j) atomicAdd(&pd[j], alpha * pl[j]);
}

// ---------------- kernel 8: bias + ELU + residual + LayerNorm (one wave per node) -------
__global__ void k_final(const float* __restrict__ agg, const float* __restrict__ x,
                        const float* __restrict__ bias, const float* __restrict__ gamma,
                        const float* __restrict__ beta, float* __restrict__ out) {
    int wave = threadIdx.x >> 5;
    int lane = threadIdx.x & 31;
    int n = blockIdx.x * (blockDim.x >> 5) + wave;
    if (n >= NN) return;

    float h[8];
    float s = 0.0f;
    #pragma unroll
    for (int j = 0; j < 8; ++j) {
        int c = lane * 8 + j;
        float v = agg[(long)n * OUTD + c] + bias[c];
        v = (v > 0.0f) ? v : (__expf(v) - 1.0f);    // ELU
        v += x[(long)n * OUTD + c];                 // residual
        h[j] = v;
        s += v;
    }
    #pragma unroll
    for (int m = 1; m < 32; m <<= 1) s += __shfl_xor(s, m, 32);
    float mu = s * (1.0f / OUTD);

    float vs = 0.0f;
    #pragma unroll
    for (int j = 0; j < 8; ++j) { float d = h[j] - mu; vs += d * d; }
    #pragma unroll
    for (int m = 1; m < 32; m <<= 1) vs += __shfl_xor(vs, m, 32);
    float rs = rsqrtf(vs * (1.0f / OUTD) + LN_EPS);

    #pragma unroll
    for (int j = 0; j < 8; ++j) {
        int c = lane * 8 + j;
        out[(long)n * OUTD + c] = (h[j] - mu) * rs * gamma[c] + beta[c];
    }
}

extern "C" void kernel_launch(void* const* d_in, const int* in_sizes, int n_in,
                              void* d_out, int out_size, void* d_ws, size_t ws_size,
                              hipStream_t stream) {
    const float* x     = (const float*)d_in[0];
    const int*   ei    = (const int*)d_in[1];
    const float* Wl    = (const float*)d_in[2];
    const float* Wr    = (const float*)d_in[3];
    const float* att   = (const float*)d_in[4];
    const float* bias  = (const float*)d_in[5];
    const float* gamma = (const float*)d_in[6];
    const float* beta  = (const float*)d_in[7];
    float* out = (float*)d_out;

    // workspace layout
    char* ws = (char*)d_ws;
    unsigned short* xb  = (unsigned short*)ws;                 // N*256 bf16
    unsigned short* Wlb = xb + (long)NN * IND;                 // 256*256
    unsigned short* Wrb = Wlb + IND * OUTD;                    // 256*256
    float* xl       = (float*)(Wrb + IND * OUTD);              // N*256 f32
    float* xr       = xl + (long)NN * OUTD;
    float* agg      = xr + (long)NN * OUTD;
    float* node_max = agg + (long)NN * OUTD;
    float* node_sum = node_max + (long)NN * HEADS;
    float* score    = node_sum + (long)NN * HEADS;             // E*4 f32 (score then ex)

    const int BS = 256;

    k_cvt_x<<<4096, BS, 0, stream>>>(x, xb);
    k_cvt_w<<<(IND * OUTD + BS - 1) / BS, BS, 0, stream>>>(Wl, Wr, Wlb, Wrb);
    k_gemm<<<(1250 * 32) / 8, BS, 0, stream>>>(xb, Wlb, Wrb, xl, xr);
    k_init<<<4096, BS, 0, stream>>>(agg, node_max, node_sum);
    k_score<<<NE / 8, BS, 0, stream>>>(ei, xl, xr, att, score, node_max);
    k_expsum<<<((long)NE * HEADS + BS - 1) / BS, BS, 0, stream>>>(ei, score, node_max, node_sum);
    k_agg<<<NE / 8, BS, 0, stream>>>(ei, xl, score, node_sum, agg);
    k_final<<<NN / 8, BS, 0, stream>>>(agg, x, bias, gamma, beta, out);
}
